// GemmaAttention_25091198943557
// MI455X (gfx1250) — matrix operand order repaired and checked by reading, not compiled, not run on hardware
//
#include <hip/hip_runtime.h>
#include <hip/hip_bf16.h>

// ---------------------------------------------------------------------------
// Gemma attention forward for MI455X (gfx1250, wave32, WMMA).
//   B=2, S=2048, HID=2048, NH=8, NKV=1, HD=256, GQA groups=8, theta=1e4
// Outputs (concatenated in d_out, fp32):
//   [0]              attn_output  [B,S,HID]      =  8,388,608 floats
//   [B*S*HID]        attn_weights [B,NH,S,S]     = 67,108,864 floats
// Pipeline: QKV proj (WMMA bf16) -> RoPE (+Q transpose) -> QK^T*scale+mask
//           (WMMA bf16, NT) -> row softmax fp32 -> PV (WMMA bf16) -> O proj.
// LDS tiles are stored in WMMA-fragment order (contiguous 32B/lane, 16B
// aligned -> 2x ds_load_b128 per fragment) and double-buffered so the
// global-load/cvt/ds-store pipeline for tile k+1 overlaps the WMMAs of tile
// k, with a single workgroup barrier per K-step.
// ---------------------------------------------------------------------------

#define B_   2
#define S_   2048
#define HID_ 2048
#define NH_  8
#define HD_  256

typedef __attribute__((ext_vector_type(16))) __bf16 v16bf;
typedef __attribute__((ext_vector_type(8)))  __bf16 v8bf;
typedef __attribute__((ext_vector_type(2)))  __bf16 v2bf;
typedef __attribute__((ext_vector_type(8)))  float  v8f;

// Load a 16-element bf16 fragment from LDS as two aligned 16-byte reads.
static __device__ __forceinline__ v16bf ldsFrag(const __bf16* p)
{
    v8bf lo = *(const v8bf*)p;
    v8bf hi = *(const v8bf*)(p + 8);
    return __builtin_shufflevector(lo, hi, 0, 1, 2, 3, 4, 5, 6, 7,
                                   8, 9, 10, 11, 12, 13, 14, 15);
}

// ---------------------------------------------------------------------------
// Generic batched GEMM:  C[z] = alpha * A[z] * op(B[z]) (+ mask[z])
//   A: row-major [M,K] (lda == K for every use here)
//   B: BT=0 -> row-major [K,N] (ldb = N);  BT=1 -> row-major [N,K] (ldb = K)
//   per-z offsets: off = (z/div)*hi + (z%div)*lo   (shared-KV GQA batching,
//   fused [b,s,h*HD] context output layout)
// Tile: BM=64, BN=128, BK=32; 8 waves; wave = 4 WMMA C-tiles sharing 1 A-frag.
// LDS: A tile [m][k] (m-major, k in natural order) so the
//      A fragment (k 0..15 / 16..31 by lane-half) is contiguous;
//      B tile [n][k] (n-major) so the B fragment (k 0..15 / 16..31 by
//      lane-half) is contiguous. Rows padded to 40 elems (80 B) for alignment
//      and bank spread. Double-buffered (2 x ~15 KB).
// ---------------------------------------------------------------------------
template<bool BT>
__global__ __launch_bounds__(256)
void gemm_wmma_bf16(const float* __restrict__ A, const float* __restrict__ Bm,
                    float* __restrict__ C, const float* __restrict__ Msk,
                    int M, int N, int K, int ldb, int ldc, float alpha,
                    long long hiA, long long loA, int divA,
                    long long hiB, long long loB, int divB,
                    long long hiC, long long loC, int divC,
                    long long hiM, int divM, int maskLd)
{
    __shared__ __align__(16) __bf16 ldsA[2][64][40];    // [buf][m][k]
    __shared__ __align__(16) __bf16 ldsB[2][128][40];   // [buf][n][k]

    const int t = threadIdx.x;
    const int z = blockIdx.z;
    const long long offA = (long long)(z / divA) * hiA + (long long)(z % divA) * loA;
    const long long offB = (long long)(z / divB) * hiB + (long long)(z % divB) * loB;
    const long long offC = (long long)(z / divC) * hiC + (long long)(z % divC) * loC;
    const long long offM = Msk ? (long long)(z / divM) * hiM : 0;

    const int m0 = blockIdx.y * 64;
    const int n0 = blockIdx.x * 128;

    const int lane = t & 31;
    const int w    = t >> 5;          // 8 waves
    const int mr   = (w >> 1) << 4;   // wave row-tile: 0,16,32,48
    const int nc   = (w & 1)  << 6;   // wave col base: 0 or 64 (4 tiles)

    const int ktiles = K >> 5;

    // Per-thread staging coordinates (loop-invariant).
    const int ar = t >> 2;                               // 0..63
    const int ac = (t & 3) << 3;                         // 0,8,16,24
    const int pg = ac;                                   // 0,8,16,24

    // Stage K-tile `kt` into LDS buffer `buf` (fp32 global -> bf16 LDS).
    auto stageTiles = [&](int buf, int kt) {
        {
            const float* ap = A + offA + (long long)(m0 + ar) * K + (kt << 5) + ac;
            if (kt + 1 < ktiles) __builtin_prefetch(ap + 32, 0, 0);
            #pragma unroll
            for (int j = 0; j < 8; ++j) ldsA[buf][ar][pg + j] = (__bf16)ap[j];
        }
        if (BT) {
            // global [N,K]: contiguous k per row -> contiguous LDS row
            const int bn = t >> 1;                       // 0..127
            const int bk = (t & 1) << 4;                 // 0 or 16
            const float* bp = Bm + offB + (long long)(n0 + bn) * ldb + (kt << 5) + bk;
            if (kt + 1 < ktiles) __builtin_prefetch(bp + 32, 0, 0);
            #pragma unroll
            for (int j = 0; j < 16; ++j) ldsB[buf][bn][bk + j] = (__bf16)bp[j];
        } else {
            // global [K,N]: read k-pair rows, pack 2 bf16 -> one b32 LDS store
            const int br = (t >> 4) << 1;                // 0,2,..,30
            const int bc = (t & 15) << 3;                // 0..120
            const float* bp0 = Bm + offB + (long long)((kt << 5) + br) * ldb + n0 + bc;
            const float* bp1 = bp0 + ldb;
            if (kt + 1 < ktiles) __builtin_prefetch(bp0 + 32LL * ldb, 0, 0);
            #pragma unroll
            for (int j = 0; j < 8; ++j) {
                v2bf pk;
                pk[0] = (__bf16)bp0[j];     // k = br
                pk[1] = (__bf16)bp1[j];     // k = br+1
                *(v2bf*)&ldsB[buf][bc + j][br] = pk;
            }
        }
    };

    v8f acc[4] = {{}, {}, {}, {}};

    // Software pipeline: stage(0) | { stage(next) || compute(cur) ; barrier }
    stageTiles(0, 0);
    __syncthreads();

    for (int kt = 0; kt < ktiles; ++kt) {
        const int cur = kt & 1;
        if (kt + 1 < ktiles) stageTiles(cur ^ 1, kt + 1);

        // ---- fragments: contiguous 32B per lane, 16B-aligned ----
        const v16bf af = ldsFrag(&ldsA[cur][mr + (lane & 15)][(lane < 16) ? 0 : 16]);
        const int kb = (lane < 16) ? 0 : 16;
        #pragma unroll
        for (int s = 0; s < 4; ++s) {
            const v16bf bf = ldsFrag(&ldsB[cur][nc + (s << 4) + (lane & 15)][kb]);
            acc[s] = __builtin_amdgcn_wmma_f32_16x16x32_bf16(
                false, af, false, bf, (short)0, acc[s], false, false);
        }
        // Publishes buffer cur^1 for iteration kt+1 AND guarantees all waves
        // are done reading buffer cur before iteration kt+1 overwrites it.
        __syncthreads();
    }

    // ---- epilogue: C/D layout -> VGPR r is rows r (lanes 0-15) / r+8 ----
    const int rowBase = m0 + mr + ((lane < 16) ? 0 : 8);
    const int col0    = n0 + nc + (lane & 15);
    #pragma unroll
    for (int s = 0; s < 4; ++s) {
        const int cc = col0 + (s << 4);
        #pragma unroll
        for (int r = 0; r < 8; ++r) {
            const int gr = rowBase + r;
            float v = acc[s][r] * alpha;
            if (Msk) v += Msk[offM + (long long)gr * maskLd + cc];
            C[offC + (long long)gr * ldc + cc] = v;
        }
    }
}

// ---------------------------------------------------------------------------
// RoPE: read Qp [B,S,NH*HD], Kp [B,S,HD]; write Qr [B,NH,S,HD] (transposed
// for the attention GEMMs) and Kr [B,S,HD]. One block per (b,s), 256 = HD.
// ---------------------------------------------------------------------------
__global__ __launch_bounds__(256)
void rope_apply(const float* __restrict__ Qp, const float* __restrict__ Kp,
                const int* __restrict__ pos,
                float* __restrict__ Qr, float* __restrict__ Kr)
{
    const int bs = blockIdx.x;            // 0 .. B*S-1
    const int b  = bs >> 11;              // /S_
    const int s  = bs & (S_ - 1);
    const int d  = threadIdx.x;           // 0..255
    const int j  = d & (HD_ / 2 - 1);     // frequency index

    // inv_freq = theta^(-2j/HD) = exp(-j * ln(1e4)/128)
    const float inv_freq = __expf(-(float)j * (9.2103403719761836f / 128.0f));
    const float ang = (float)pos[bs] * inv_freq;
    float sn, cs;
    __sincosf(ang, &sn, &cs);

    // K (single KV head), already in [b,s,d] layout
    {
        const float* kin = Kp + (long long)bs * HD_;
        const float kv = kin[d];
        const float kr = (d < HD_ / 2) ? -kin[d + HD_ / 2] : kin[d - HD_ / 2];
        Kr[(long long)bs * HD_ + d] = kv * cs + kr * sn;
    }
    // Q: all heads, transpose to [b,h,s,d]
    #pragma unroll
    for (int h = 0; h < NH_; ++h) {
        const float* qin = Qp + (long long)bs * (NH_ * HD_) + h * HD_;
        const float qv = qin[d];
        const float qr = (d < HD_ / 2) ? -qin[d + HD_ / 2] : qin[d - HD_ / 2];
        Qr[(((long long)b * NH_ + h) * S_ + s) * HD_ + d] = qv * cs + qr * sn;
    }
}

// ---------------------------------------------------------------------------
// In-place fp32 row softmax over the attn_weights region. One block per row
// (grid = [S, B*NH]); 256 threads x 8 elements, wave32 shfl + LDS reduce.
// ---------------------------------------------------------------------------
__global__ __launch_bounds__(256)
void softmax_rows(float* __restrict__ attn)
{
    const long long row = (long long)blockIdx.y * S_ + blockIdx.x;
    float* p = attn + row * S_;
    const int t    = threadIdx.x;
    const int lane = t & 31;
    const int w    = t >> 5;

    __shared__ float redA[8];
    __shared__ float redB[8];

    float v[8];
    float mx = -3.4e38f;
    #pragma unroll
    for (int j = 0; j < 8; ++j) { v[j] = p[t + (j << 8)]; mx = fmaxf(mx, v[j]); }
    #pragma unroll
    for (int o = 16; o; o >>= 1) mx = fmaxf(mx, __shfl_xor(mx, o, 32));
    if (lane == 0) redA[w] = mx;
    __syncthreads();
    if (t == 0) {
        float m = redA[0];
        #pragma unroll
        for (int i = 1; i < 8; ++i) m = fmaxf(m, redA[i]);
        redA[0] = m;
    }
    __syncthreads();
    mx = redA[0];

    float sum = 0.0f;
    #pragma unroll
    for (int j = 0; j < 8; ++j) { v[j] = __expf(v[j] - mx); sum += v[j]; }
    #pragma unroll
    for (int o = 16; o; o >>= 1) sum += __shfl_xor(sum, o, 32);
    if (lane == 0) redB[w] = sum;
    __syncthreads();
    if (t == 0) {
        float s = 0.0f;
        #pragma unroll
        for (int i = 0; i < 8; ++i) s += redB[i];
        redB[0] = s;
    }
    __syncthreads();
    const float inv = 1.0f / redB[0];
    #pragma unroll
    for (int j = 0; j < 8; ++j) p[t + (j << 8)] = v[j] * inv;
}

// ---------------------------------------------------------------------------
extern "C" void kernel_launch(void* const* d_in, const int* in_sizes, int n_in,
                              void* d_out, int out_size, void* d_ws, size_t ws_size,
                              hipStream_t stream)
{
    (void)in_sizes; (void)n_in; (void)out_size; (void)ws_size;

    const float* hidden = (const float*)d_in[0];
    const float* amask  = (const float*)d_in[1];   // [B,1,S,S]
    const int*   pos    = (const int*)d_in[2];     // [B,S] int32
    const float* Wq     = (const float*)d_in[3];   // [HID, NH*HD]
    const float* Wk     = (const float*)d_in[4];   // [HID, HD]
    const float* Wv     = (const float*)d_in[5];   // [HID, HD]
    const float* Wo     = (const float*)d_in[6];   // [NH*HD, HID]

    float* out   = (float*)d_out;                         // [B,S,HID]
    float* attnW = out + (long long)B_ * S_ * HID_;       // [B,NH,S,S]

    // Workspace layout (fp32). Ctx aliases Qp (dead after RoPE). ~80 MB.
    float* Qp  = (float*)d_ws;                            // [B*S, NH*HD]
    float* Kp  = Qp + (long long)B_ * S_ * HID_;          // [B*S, HD]
    float* Vp  = Kp + (long long)B_ * S_ * HD_;           // [B*S, HD]
    float* Qr  = Vp + (long long)B_ * S_ * HD_;           // [B,NH,S,HD]
    float* Kr  = Qr + (long long)B_ * NH_ * S_ * HD_;     // [B,S,HD]
    float* Ctx = Qp;                                      // [B,S,NH*HD]

    const dim3 blk(256);
    const long long SHD = (long long)S_ * HD_;    // 524288
    const long long SS  = (long long)S_ * S_;     // 4194304
    const long long SH  = (long long)S_ * HID_;   // 4194304

    // 1) Q/K/V projections: [4096,2048] x [2048,N]
    gemm_wmma_bf16<false><<<dim3(HID_ / 128, (B_ * S_) / 64, 1), blk, 0, stream>>>(
        hidden, Wq, Qp, nullptr, B_ * S_, HID_, HID_, HID_, HID_, 1.0f,
        0, 0, 1,  0, 0, 1,  0, 0, 1,  0, 1, 0);
    gemm_wmma_bf16<false><<<dim3(HD_ / 128, (B_ * S_) / 64, 1), blk, 0, stream>>>(
        hidden, Wk, Kp, nullptr, B_ * S_, HD_, HID_, HD_, HD_, 1.0f,
        0, 0, 1,  0, 0, 1,  0, 0, 1,  0, 1, 0);
    gemm_wmma_bf16<false><<<dim3(HD_ / 128, (B_ * S_) / 64, 1), blk, 0, stream>>>(
        hidden, Wv, Vp, nullptr, B_ * S_, HD_, HID_, HD_, HD_, 1.0f,
        0, 0, 1,  0, 0, 1,  0, 0, 1,  0, 1, 0);

    // 2) RoPE (+ Q transpose to [b,h,s,d])
    rope_apply<<<dim3(B_ * S_), blk, 0, stream>>>(Qp, Kp, pos, Qr, Kr);

    // 3) scores = Q K^T / sqrt(HD) + mask   (NT GEMM, z = (b,h), KV shared per b)
    gemm_wmma_bf16<true><<<dim3(S_ / 128, S_ / 64, B_ * NH_), blk, 0, stream>>>(
        Qr, Kr, attnW, amask, S_, S_, HD_, HD_, S_, 0.0625f,
        SHD, 0, 1,        // A: z * S*HD
        SHD, 0, NH_,      // B: (z/NH) * S*HD   (shared KV)
        SS,  0, 1,        // C: z * S*S
        SS, NH_, S_);     // mask: (z/NH) * S*S

    // 4) row softmax (fp32, in place on attn_weights output)
    softmax_rows<<<dim3(S_, B_ * NH_), blk, 0, stream>>>(attnW);

    // 5) ctx = attn @ V, written directly in fused [b, s, h*HD] layout
    gemm_wmma_bf16<false><<<dim3(HD_ / 128, S_ / 64, B_ * NH_), blk, 0, stream>>>(
        attnW, Vp, Ctx, nullptr, S_, HD_, S_, HD_, HID_, 1.0f,
        SS,  0, 1,        // A: z * S*S
        SHD, 0, NH_,      // B: (z/NH) * S*HD   (shared V)
        SH, 256, NH_,     // C: (z/NH)*S*HID + (z%NH)*HD
        0, 1, 0);

    // 6) attn_output = ctx @ Wo : [4096,2048] x [2048,2048]
    gemm_wmma_bf16<false><<<dim3(HID_ / 128, (B_ * S_) / 64, 1), blk, 0, stream>>>(
        Ctx, Wo, out, nullptr, B_ * S_, HID_, HID_, HID_, HID_, 1.0f,
        0, 0, 1,  0, 0, 1,  0, 0, 1,  0, 1, 0);
}
